// FusedMoE_12412455485616
// MI455X (gfx1250) — matrix operand order repaired and checked by reading, not compiled, not run on hardware
//
#include <hip/hip_runtime.h>
#include <cstdint>

// ---------------------------------------------------------------------------
// Fused MoE for MI455X (gfx1250, wave32, WMMA + async-LDS copies).
//   tokens T=8192, hidden H=2048, inter I=1408, experts E=8, top_k=2
// Compute-bound (~2.8e11 routed FLOP; weights 277MB = 12us @ 23.3TB/s), so:
//  - route top-2 only (compacted per-expert token lists)
//  - bf16 WMMA math; weights pre-converted to bf16 once per call in ws
//  - per-expert weight tiles re-read from the 192MB L2
//  - LDS staging via GLOBAL_LOAD_ASYNC_TO_LDS_B128 (ASYNCcnt), double-buffered
//    so copies for K-step k+1 overlap WMMAs of K-step k.
// Workspace requirement: ~187 MB.
// ---------------------------------------------------------------------------

#define T_TOK 8192
#define HDIM  2048
#define IDIM  1408
#define NEXP  8

typedef __attribute__((ext_vector_type(16))) __bf16 v16bf;
typedef __attribute__((ext_vector_type(8)))  float  v8f;
typedef __attribute__((ext_vector_type(4)))  float  f32x4;
typedef __attribute__((ext_vector_type(4)))  unsigned int u32x4;
typedef __attribute__((ext_vector_type(2)))  unsigned int u32x2;

union FragBF { v16bf v; u32x4 q[2]; };

__device__ __forceinline__ unsigned short f2bf(float f) {
  unsigned int u = __float_as_uint(f);
  u += 0x7FFFu + ((u >> 16) & 1u);   // round-to-nearest-even
  return (unsigned short)(u >> 16);
}

// Low 32 bits of a generic LDS pointer == LDS byte offset (aperture is in the
// high bits per the CDNA5 aperture check), which is what async-LDS VDST wants.
__device__ __forceinline__ unsigned lds_addr(const void* p) {
  return (unsigned)(unsigned long long)(uintptr_t)p;
}

// Async 16B global -> LDS copy, tracked by ASYNCcnt (no VGPR round-trip).
__device__ __forceinline__ void async_cp16(unsigned lds, unsigned long long g) {
  asm volatile("global_load_async_to_lds_b128 %0, %1, off"
               :: "v"(lds), "v"(g) : "memory");
}
__device__ __forceinline__ void wait_async0() {
  asm volatile("s_wait_asynccnt 0x0" ::: "memory");
}

// ---------------------------------------------------------------------------
// Zero output accumulator + per-expert counters (every call; harness poisons).
// ---------------------------------------------------------------------------
__global__ __launch_bounds__(256) void moe_init(float* __restrict__ out,
                                                int* __restrict__ counts) {
  size_t idx = ((size_t)blockIdx.x * 256 + threadIdx.x) * 4;
  f32x4 z = {0.f, 0.f, 0.f, 0.f};
  *(f32x4*)(out + idx) = z;
  if (blockIdx.x == 0 && threadIdx.x < NEXP) counts[threadIdx.x] = 0;
}

// ---------------------------------------------------------------------------
// fp32 -> bf16 bulk cast (used for x, w13, w2).
// ---------------------------------------------------------------------------
__global__ __launch_bounds__(256) void moe_cast(const float* __restrict__ src,
                                                unsigned short* __restrict__ dst) {
  size_t idx = ((size_t)blockIdx.x * 256 + threadIdx.x) * 4;
  f32x4 f = *(const f32x4*)(src + idx);
  u32x2 p = { (unsigned)f2bf(f[0]) | ((unsigned)f2bf(f[1]) << 16),
              (unsigned)f2bf(f[2]) | ((unsigned)f2bf(f[3]) << 16) };
  *(u32x2*)(dst + idx) = p;
}

// ---------------------------------------------------------------------------
// Router: one wave32 per token.
// ---------------------------------------------------------------------------
__global__ __launch_bounds__(256) void moe_router(
    const float* __restrict__ x, const float* __restrict__ gw,
    float* __restrict__ logits, float* __restrict__ ce,
    int* __restrict__ counts, int* __restrict__ tok) {
  __shared__ float g[NEXP * HDIM];          // 64 KB of 320 KB LDS
  const int tid = threadIdx.x;
  for (int i = tid; i < NEXP * HDIM; i += 256) g[i] = gw[i];
  __syncthreads();

  const int lane = tid & 31;
  const int wave = tid >> 5;
  const int t = blockIdx.x * 8 + wave;
  const float* xr = x + (size_t)t * HDIM;

  float acc[NEXP];
#pragma unroll
  for (int e = 0; e < NEXP; ++e) acc[e] = 0.f;
  for (int h = lane; h < HDIM; h += 32) {
    float xv = xr[h];
#pragma unroll
    for (int e = 0; e < NEXP; ++e) acc[e] += xv * g[e * HDIM + h];
  }
#pragma unroll
  for (int e = 0; e < NEXP; ++e) {
#pragma unroll
    for (int off = 16; off >= 1; off >>= 1)
      acc[e] += __shfl_xor(acc[e], off, 32);
  }

  if (lane == 0) {
    float m = acc[0];
#pragma unroll
    for (int e = 1; e < NEXP; ++e) m = fmaxf(m, acc[e]);
    float p[NEXP];
#pragma unroll
    for (int e = 0; e < NEXP; ++e) p[e] = __expf(acc[e] - m);
    int i1 = 0;
#pragma unroll
    for (int e = 1; e < NEXP; ++e) if (p[e] > p[i1]) i1 = e;
    int i2 = (i1 == 0) ? 1 : 0;
#pragma unroll
    for (int e = 0; e < NEXP; ++e) if (e != i1 && p[e] > p[i2]) i2 = e;
    float wsum = p[i1] + p[i2];
    float c1 = p[i1] / wsum, c2 = p[i2] / wsum;
#pragma unroll
    for (int e = 0; e < NEXP; ++e) {
      logits[(size_t)t * NEXP + e] = acc[e];
      ce[(size_t)e * T_TOK + t] = 0.f;
    }
    ce[(size_t)i1 * T_TOK + t] = c1;
    ce[(size_t)i2 * T_TOK + t] = c2;
    int q1 = atomicAdd(&counts[i1], 1); tok[(size_t)i1 * T_TOK + q1] = t;
    int q2 = atomicAdd(&counts[i2], 1); tok[(size_t)i2 * T_TOK + q2] = t;
  }
}

// ---------------------------------------------------------------------------
// GEMM1: h = silu(x @ w1e^T) * (x @ w3e^T), bf16 WMMA, async double-buffered
// LDS staging. Block = 128 threads = 4 wave32s, tile 64(M)x64(N), K-step 32.
// ---------------------------------------------------------------------------
__global__ __launch_bounds__(128) void moe_gemm1(
    const unsigned short* __restrict__ xb, const unsigned short* __restrict__ w13b,
    const int* __restrict__ counts, const int* __restrict__ tok,
    unsigned short* __restrict__ hb, int e) {
  const int cnt  = counts[e];
  const int row0 = blockIdx.y * 64;
  if (row0 >= cnt) return;                      // uniform block early-exit
  const int n0 = blockIdx.x * 64;
  const unsigned short* w1 = w13b + (size_t)e * (2 * IDIM) * HDIM + (size_t)n0 * HDIM;
  const unsigned short* w3 = w1 + (size_t)IDIM * HDIM;
  const int* tl = tok + (size_t)e * T_TOK + row0;

  __shared__ __align__(16) unsigned short sA [2][64 * 32];
  __shared__ __align__(16) unsigned short sB1[2][64 * 32];
  __shared__ __align__(16) unsigned short sB3[2][64 * 32];
  __shared__ int stok[64];

  const int tid = threadIdx.x;
  if (tid < 64) stok[tid] = (row0 + tid < cnt) ? tl[tid] : tl[0];
  __syncthreads();

  // Per-thread staging assignment: 16B chunks, rows r0 and r0+32, col c0.
  const int r0 = tid >> 2;
  const int c0 = (tid & 3) * 8;
  const int tok0 = stok[r0], tok1 = stok[r0 + 32];
  const unsigned long long gA0 = (unsigned long long)(uintptr_t)(xb + (size_t)tok0 * HDIM + c0);
  const unsigned long long gA1 = (unsigned long long)(uintptr_t)(xb + (size_t)tok1 * HDIM + c0);
  const unsigned long long gB1a = (unsigned long long)(uintptr_t)(w1 + (size_t)r0 * HDIM + c0);
  const unsigned long long gB1b = (unsigned long long)(uintptr_t)(w1 + (size_t)(r0 + 32) * HDIM + c0);
  const unsigned long long gB3a = (unsigned long long)(uintptr_t)(w3 + (size_t)r0 * HDIM + c0);
  const unsigned long long gB3b = (unsigned long long)(uintptr_t)(w3 + (size_t)(r0 + 32) * HDIM + c0);
  unsigned lA0[2], lA1[2], lB1a[2], lB1b[2], lB3a[2], lB3b[2];
#pragma unroll
  for (int st = 0; st < 2; ++st) {
    lA0 [st] = lds_addr(&sA [st][r0 * 32 + c0]);
    lA1 [st] = lds_addr(&sA [st][(r0 + 32) * 32 + c0]);
    lB1a[st] = lds_addr(&sB1[st][r0 * 32 + c0]);
    lB1b[st] = lds_addr(&sB1[st][(r0 + 32) * 32 + c0]);
    lB3a[st] = lds_addr(&sB3[st][r0 * 32 + c0]);
    lB3b[st] = lds_addr(&sB3[st][(r0 + 32) * 32 + c0]);
  }
  auto issue = [&](int st, int k0) {
    unsigned long long b = (unsigned long long)(2 * k0);   // bytes
    async_cp16(lA0 [st], gA0  + b);
    async_cp16(lA1 [st], gA1  + b);
    async_cp16(lB1a[st], gB1a + b);
    async_cp16(lB1b[st], gB1b + b);
    async_cp16(lB3a[st], gB3a + b);
    async_cp16(lB3b[st], gB3b + b);
  };

  const int lane = tid & 31;
  const int wave = tid >> 5;
  const int wm = (wave & 1) * 32;
  const int wn = (wave >> 1) * 32;
  const int aRow = lane & 15;
  const int aOff = (lane < 16) ? 0 : 8;    // A frag: K chunks {0,16}/{8,24}
  const int bOff = (lane < 16) ? 0 : 8;    // B frag: K chunks {0,16}/{8,24}

  v8f acc1[2][2] = {};
  v8f acc3[2][2] = {};

  issue(0, 0);
  for (int k0 = 0; k0 < HDIM; k0 += 32) {
    const int st = (k0 >> 5) & 1;
    wait_async0();                 // my stage-st copies landed in LDS
    __syncthreads();               // everyone's landed; prev reads retired
    if (k0 + 32 < HDIM) issue(st ^ 1, k0 + 32);   // overlap with WMMAs below

    FragBF af[2], b1f[2], b3f[2];
#pragma unroll
    for (int fm = 0; fm < 2; ++fm) {
      int r = wm + fm * 16 + aRow;
      af[fm].q[0] = *(const u32x4*)&sA[st][r * 32 + aOff];
      af[fm].q[1] = *(const u32x4*)&sA[st][r * 32 + aOff + 16];
    }
#pragma unroll
    for (int fn = 0; fn < 2; ++fn) {
      int n = wn + fn * 16 + aRow;
      b1f[fn].q[0] = *(const u32x4*)&sB1[st][n * 32 + bOff];
      b1f[fn].q[1] = *(const u32x4*)&sB1[st][n * 32 + bOff + 16];
      b3f[fn].q[0] = *(const u32x4*)&sB3[st][n * 32 + bOff];
      b3f[fn].q[1] = *(const u32x4*)&sB3[st][n * 32 + bOff + 16];
    }
#pragma unroll
    for (int fm = 0; fm < 2; ++fm) {
#pragma unroll
      for (int fn = 0; fn < 2; ++fn) {
        acc1[fm][fn] = __builtin_amdgcn_wmma_f32_16x16x32_bf16(
            false, af[fm].v, false, b1f[fn].v, (short)0, acc1[fm][fn], false, false);
        acc3[fm][fn] = __builtin_amdgcn_wmma_f32_16x16x32_bf16(
            false, af[fm].v, false, b3f[fn].v, (short)0, acc3[fm][fn], false, false);
      }
    }
  }

  // epilogue: silu(a1)*a3 -> bf16 h rows (compacted positions)
  const int mBase = (lane >= 16) ? 8 : 0;
  const int nLane = lane & 15;
#pragma unroll
  for (int fm = 0; fm < 2; ++fm) {
#pragma unroll
    for (int fn = 0; fn < 2; ++fn) {
#pragma unroll
      for (int i = 0; i < 8; ++i) {
        int row = row0 + wm + fm * 16 + mBase + i;
        if (row < cnt) {
          float a1 = acc1[fm][fn][i];
          float a3 = acc3[fm][fn][i];
          float hv = (a1 / (1.f + __expf(-a1))) * a3;
          hb[(size_t)row * IDIM + n0 + wn + fn * 16 + nLane] = f2bf(hv);
        }
      }
    }
  }
}

// ---------------------------------------------------------------------------
// GEMM2: out[tok] += ce * (h @ w2e^T). Scatter accumulate; experts serialized
// on the stream, each token appears at most once per expert -> no races.
// ---------------------------------------------------------------------------
__global__ __launch_bounds__(128) void moe_gemm2(
    const unsigned short* __restrict__ hb, const unsigned short* __restrict__ w2b,
    const float* __restrict__ ce, const int* __restrict__ counts,
    const int* __restrict__ tok, float* __restrict__ out, int e) {
  const int cnt  = counts[e];
  const int row0 = blockIdx.y * 64;
  if (row0 >= cnt) return;
  const int n0 = blockIdx.x * 64;
  const unsigned short* wB = w2b + (size_t)e * HDIM * IDIM + (size_t)n0 * IDIM;
  const int* tl = tok + (size_t)e * T_TOK + row0;

  __shared__ __align__(16) unsigned short sA[2][64 * 32];
  __shared__ __align__(16) unsigned short sB[2][64 * 32];
  __shared__ int stok[64];

  const int tid = threadIdx.x;
  if (tid < 64) stok[tid] = (row0 + tid < cnt) ? tl[tid] : tl[0];
  __syncthreads();

  const int r0 = tid >> 2;
  const int c0 = (tid & 3) * 8;
  const unsigned long long gA0 = (unsigned long long)(uintptr_t)(hb + (size_t)(row0 + r0) * IDIM + c0);
  const unsigned long long gA1 = (unsigned long long)(uintptr_t)(hb + (size_t)(row0 + r0 + 32) * IDIM + c0);
  const unsigned long long gB0 = (unsigned long long)(uintptr_t)(wB + (size_t)r0 * IDIM + c0);
  const unsigned long long gB1 = (unsigned long long)(uintptr_t)(wB + (size_t)(r0 + 32) * IDIM + c0);
  unsigned lA0[2], lA1[2], lB0[2], lB1[2];
#pragma unroll
  for (int st = 0; st < 2; ++st) {
    lA0[st] = lds_addr(&sA[st][r0 * 32 + c0]);
    lA1[st] = lds_addr(&sA[st][(r0 + 32) * 32 + c0]);
    lB0[st] = lds_addr(&sB[st][r0 * 32 + c0]);
    lB1[st] = lds_addr(&sB[st][(r0 + 32) * 32 + c0]);
  }
  auto issue = [&](int st, int k0) {
    unsigned long long b = (unsigned long long)(2 * k0);
    async_cp16(lA0[st], gA0 + b);
    async_cp16(lA1[st], gA1 + b);
    async_cp16(lB0[st], gB0 + b);
    async_cp16(lB1[st], gB1 + b);
  };

  const int lane = tid & 31;
  const int wave = tid >> 5;
  const int wm = (wave & 1) * 32;
  const int wn = (wave >> 1) * 32;
  const int aRow = lane & 15;
  const int aOff = (lane < 16) ? 0 : 8;
  const int bOff = (lane < 16) ? 0 : 8;

  v8f acc[2][2] = {};

  issue(0, 0);
  for (int k0 = 0; k0 < IDIM; k0 += 32) {
    const int st = (k0 >> 5) & 1;
    wait_async0();
    __syncthreads();
    if (k0 + 32 < IDIM) issue(st ^ 1, k0 + 32);

    FragBF af[2], bf[2];
#pragma unroll
    for (int fm = 0; fm < 2; ++fm) {
      int r = wm + fm * 16 + aRow;
      af[fm].q[0] = *(const u32x4*)&sA[st][r * 32 + aOff];
      af[fm].q[1] = *(const u32x4*)&sA[st][r * 32 + aOff + 16];
    }
#pragma unroll
    for (int fn = 0; fn < 2; ++fn) {
      int n = wn + fn * 16 + aRow;
      bf[fn].q[0] = *(const u32x4*)&sB[st][n * 32 + bOff];
      bf[fn].q[1] = *(const u32x4*)&sB[st][n * 32 + bOff + 16];
    }
#pragma unroll
    for (int fm = 0; fm < 2; ++fm) {
#pragma unroll
      for (int fn = 0; fn < 2; ++fn) {
        acc[fm][fn] = __builtin_amdgcn_wmma_f32_16x16x32_bf16(
            false, af[fm].v, false, bf[fn].v, (short)0, acc[fm][fn], false, false);
      }
    }
  }

  const int mBase = (lane >= 16) ? 8 : 0;
  const int nLane = lane & 15;
#pragma unroll
  for (int fm = 0; fm < 2; ++fm) {
#pragma unroll
    for (int fn = 0; fn < 2; ++fn) {
#pragma unroll
      for (int i = 0; i < 8; ++i) {
        int lrow = wm + fm * 16 + mBase + i;
        int row = row0 + lrow;
        if (row < cnt) {
          int t = stok[lrow];
          float c = ce[(size_t)e * T_TOK + t];
          size_t o = (size_t)t * HDIM + n0 + wn + fn * 16 + nLane;
          out[o] += c * acc[fm][fn][i];
        }
      }
    }
  }
}

// ---------------------------------------------------------------------------
// Host driver: all launches on `stream`, graph-capture safe, deterministic.
// ---------------------------------------------------------------------------
extern "C" void kernel_launch(void* const* d_in, const int* in_sizes, int n_in,
                              void* d_out, int out_size, void* d_ws, size_t ws_size,
                              hipStream_t stream) {
  (void)in_sizes; (void)n_in; (void)out_size; (void)ws_size;
  const float* x   = (const float*)d_in[0];   // (4,2048,2048) fp32
  const float* gw  = (const float*)d_in[1];   // (8,2048)
  const float* w13 = (const float*)d_in[2];   // (8,2816,2048)
  const float* w2  = (const float*)d_in[3];   // (8,2048,1408)
  // d_in[4] = top_k (=2), hardcoded

  float* out    = (float*)d_out;                         // T*H
  float* logits = out + (size_t)T_TOK * HDIM;            // T*E

  // workspace layout (~187 MB)
  char* ws = (char*)d_ws;
  float* ce   = (float*)ws;                                        // T*E f32
  int* counts = (int*)(ws + (size_t)T_TOK * NEXP * 4);             // 8 ints
  int* tokL   = counts + 64;                                       // E*T ints
  unsigned short* xb   = (unsigned short*)(tokL + (size_t)NEXP * T_TOK); // T*H
  unsigned short* hb   = xb + (size_t)T_TOK * HDIM;                // T*I
  unsigned short* w13b = hb + (size_t)T_TOK * IDIM;                // E*2I*H
  unsigned short* w2b  = w13b + (size_t)NEXP * 2 * IDIM * HDIM;    // E*H*I

  moe_init<<<(T_TOK * HDIM) / 1024, 256, 0, stream>>>(out, counts);
  moe_cast<<<(T_TOK * HDIM) / 1024, 256, 0, stream>>>(x, xb);
  moe_cast<<<(NEXP * 2 * IDIM * HDIM) / 1024, 256, 0, stream>>>(w13, w13b);
  moe_cast<<<(NEXP * HDIM * IDIM) / 1024, 256, 0, stream>>>(w2, w2b);
  moe_router<<<T_TOK / 8, 256, 0, stream>>>(x, gw, logits, ce, counts, tokL);

  dim3 g1(IDIM / 64, T_TOK / 64);   // (22, 128); tiles past count[e] early-exit
  dim3 g2(HDIM / 64, T_TOK / 64);   // (32, 128)
  for (int e = 0; e < NEXP; ++e) {
    moe_gemm1<<<g1, 128, 0, stream>>>(xb, w13b, counts, tokL, hb, e);
    moe_gemm2<<<g2, 128, 0, stream>>>(hb, w2b, ce, counts, tokL, out, e);
  }
}
